// InfrastructureGNN_79345225826602
// MI455X (gfx1250) — compile-verified, matrix operand order checked by reading, not run on hardware
//
#include <hip/hip_runtime.h>
#include <math.h>

typedef __attribute__((ext_vector_type(16))) _Float16     v16h;
typedef __attribute__((ext_vector_type(8)))  _Float16     v8h;
typedef __attribute__((ext_vector_type(4)))  _Float16     v4h;
typedef __attribute__((ext_vector_type(8)))  float        v8f;

#define F_DIM  32
#define H_DIM  128
#define HEADS  4
#define DH     128
#define QKV    (HEADS * DH)   // 512
#define LAYERS 2

// ---------------- helpers ----------------

__device__ __forceinline__ float warp_sum(float v) {
#pragma unroll
  for (int m = 16; m > 0; m >>= 1) v += __shfl_xor(v, m, 32);
  return v;
}

__device__ __forceinline__ void atomicMaxF(float* addr, float val) {
  if (val >= 0.0f) atomicMax((int*)addr, __float_as_int(val));
  else             atomicMin((unsigned int*)addr, __float_as_uint(val));
}

__device__ __forceinline__ float gelu_exact(float x) {
  return 0.5f * x * (1.0f + erff(x * 0.70710678118654752f));
}

// ---------------- f32 -> f16 conversion ----------------
__global__ void cvt_h_k(const float* __restrict__ src, _Float16* __restrict__ dst, int n) {
  int t = blockIdx.x * blockDim.x + threadIdx.x;
  if (t < n) dst[t] = (_Float16)src[t];
}

// ---------------- WMMA GEMM:  C[M,Nc] = act(A[M,K] @ B[K,Nc] + bias) ----------------
// A,B f16.  256 thr = 8 waves; block tile 128x64; wave tile 16x64 (4 accs).
// All fragments preloaded, then 4 back-to-back WMMAs per K-step.
__global__ __launch_bounds__(256)
void gemm_f16_wmma(const _Float16* __restrict__ A, int lda,
                   const _Float16* __restrict__ B, int ldb,
                   const float* __restrict__ bias,
                   float* __restrict__ Cf, _Float16* __restrict__ Ch, int ldc,
                   int M, int K, int act)
{
  __shared__ _Float16 As[128 * 32];   // [row][k]
  __shared__ _Float16 Bs[64 * 32];    // transposed: [n][k]

  const int t    = threadIdx.x;
  const int lane = t & 31;
  const int wid  = t >> 5;       // 0..7 : 16-row strip
  const int half = lane >> 4;    // 0/1
  const int l16  = lane & 15;
  const int brow = blockIdx.x * 128;
  const int bcol = blockIdx.y * 64;

  v8f acc[4];
#pragma unroll
  for (int i = 0; i < 4; ++i) acc[i] = (v8f){};

  // staging addresses (unconditional; clamped rows only feed discarded C rows)
  const int ar0 = (t >> 2);             // 0..63
  const int ac0 = (t & 3) * 8;          // 0,8,16,24
  const int gr0 = min(brow + ar0,       M - 1);
  const int gr1 = min(brow + ar0 + 64,  M - 1);
  const int bk  = t >> 3;               // 0..31
  const int bn8 = (t & 7) * 8;          // 0..56

  union Frag { v16h h; v8h p[2]; };

  for (int k0 = 0; k0 < K; k0 += 32) {
    // ---- issue all staging loads, then store to LDS ----
    v8h a0 = *(const v8h*)&A[(size_t)gr0 * lda + k0 + ac0];
    v8h a1 = *(const v8h*)&A[(size_t)gr1 * lda + k0 + ac0];
    v8h bb = *(const v8h*)&B[(size_t)(k0 + bk) * ldb + bcol + bn8];
    if (k0 + 32 < K) {
      __builtin_prefetch(&A[(size_t)gr0 * lda + k0 + 32 + ac0], 0, 1);
      __builtin_prefetch(&B[(size_t)(k0 + 32 + bk) * ldb + bcol + bn8], 0, 1);
    }
    *(v8h*)&As[ar0 * 32 + ac0]        = a0;
    *(v8h*)&As[(ar0 + 64) * 32 + ac0] = a1;
#pragma unroll
    for (int j = 0; j < 8; ++j) Bs[(bn8 + j) * 32 + bk] = bb[j];
    __syncthreads();

    // ---- preload ALL fragments (one dscnt wait), then 4 WMMAs back-to-back ----
    Frag af, bf[4];
    const int am = wid * 16 + l16;
    af.p[0] = *(const v8h*)&As[am * 32 + 8 * half];
    af.p[1] = *(const v8h*)&As[am * 32 + 16 + 8 * half];
#pragma unroll
    for (int ti = 0; ti < 4; ++ti) {
      const int bn = ti * 16 + l16;
      bf[ti].p[0] = *(const v8h*)&Bs[bn * 32 + 8 * half];
      bf[ti].p[1] = *(const v8h*)&Bs[bn * 32 + 16 + 8 * half];
    }
#pragma unroll
    for (int ti = 0; ti < 4; ++ti) {
      acc[ti] = __builtin_amdgcn_wmma_f32_16x16x32_f16(false, af.h, false, bf[ti].h,
                                                       (short)0, acc[ti], false, false);
    }
    __syncthreads();
  }

  // ---- epilogue ----
#pragma unroll
  for (int ti = 0; ti < 4; ++ti) {
#pragma unroll
    for (int r = 0; r < 8; ++r) {
      int row = brow + wid * 16 + half * 8 + r;
      if (row >= M) continue;
      int col = bcol + ti * 16 + l16;
      float v = acc[ti][r];
      if (bias) v += bias[col];
      if (act == 1) v = gelu_exact(v);
      if (Cf) Cf[(size_t)row * ldc + col] = v;
      if (Ch) Ch[(size_t)row * ldc + col] = (_Float16)v;
    }
  }
}

// ---------------- layernorm (wave per node, H=128): f32 in, f16 out ----------------
__global__ __launch_bounds__(256)
void layernorm_h_k(const float* __restrict__ X, const float* __restrict__ s,
                   const float* __restrict__ b, _Float16* __restrict__ Y, int n)
{
  int lane = threadIdx.x & 31;
  int node = blockIdx.x * 8 + (threadIdx.x >> 5);
  if (node >= n) return;
  const float4 xv = *(const float4*)&X[(size_t)node * H_DIM + lane * 4];
  float mean = warp_sum(xv.x + xv.y + xv.z + xv.w) * (1.0f / H_DIM);
  float dx = xv.x - mean, dy = xv.y - mean, dz = xv.z - mean, dw = xv.w - mean;
  float var = warp_sum(dx * dx + dy * dy + dz * dz + dw * dw) * (1.0f / H_DIM);
  float inv = rsqrtf(var + 1e-5f);
  const float4 sv = *(const float4*)&s[lane * 4];
  const float4 bv = *(const float4*)&b[lane * 4];
  v4h o;
  o[0] = (_Float16)(dx * inv * sv.x + bv.x);
  o[1] = (_Float16)(dy * inv * sv.y + bv.y);
  o[2] = (_Float16)(dz * inv * sv.z + bv.z);
  o[3] = (_Float16)(dw * inv * sv.w + bv.w);
  *(v4h*)&Y[(size_t)node * H_DIM + lane * 4] = o;
}

// ---------------- layernorm: f32 in, f32 out (final) ----------------
__global__ __launch_bounds__(256)
void layernorm_f_k(const float* __restrict__ X, const float* __restrict__ s,
                   const float* __restrict__ b, float* __restrict__ Y, int n)
{
  int lane = threadIdx.x & 31;
  int node = blockIdx.x * 8 + (threadIdx.x >> 5);
  if (node >= n) return;
  const float4 xv = *(const float4*)&X[(size_t)node * H_DIM + lane * 4];
  float mean = warp_sum(xv.x + xv.y + xv.z + xv.w) * (1.0f / H_DIM);
  float dx = xv.x - mean, dy = xv.y - mean, dz = xv.z - mean, dw = xv.w - mean;
  float var = warp_sum(dx * dx + dy * dy + dz * dz + dw * dw) * (1.0f / H_DIM);
  float inv = rsqrtf(var + 1e-5f);
  const float4 sv = *(const float4*)&s[lane * 4];
  const float4 bv = *(const float4*)&b[lane * 4];
  float4 o;
  o.x = dx * inv * sv.x + bv.x;
  o.y = dy * inv * sv.y + bv.y;
  o.z = dz * inv * sv.z + bv.z;
  o.w = dw * inv * sv.w + bv.w;
  *(float4*)&Y[(size_t)node * H_DIM + lane * 4] = o;
}

// ---------------- fill ----------------
__global__ void fill_k(float* __restrict__ p, float v, int n) {
  int t = blockIdx.x * blockDim.x + threadIdx.x;
  if (t < n) p[t] = v;
}

// ---------------- edge attention logits + running max ----------------
__global__ __launch_bounds__(256)
void edge_logits_k(const int* __restrict__ ei, const float* __restrict__ q,
                   const float* __restrict__ k, float* __restrict__ alpha,
                   float* __restrict__ amax, int E)
{
  int t = blockIdx.x * 256 + threadIdx.x;
  if (t >= E * HEADS) return;
  int e = t >> 2, hd = t & 3;
  int src = ei[e], dst = ei[E + e];
  const float4* qp = (const float4*)&q[(size_t)dst * QKV + hd * DH];
  const float4* kp = (const float4*)&k[(size_t)src * QKV + hd * DH];
  float s = 0.0f;
#pragma unroll
  for (int i = 0; i < DH / 4; ++i) {
    float4 a = qp[i], b = kp[i];
    s += a.x * b.x + a.y * b.y + a.z * b.z + a.w * b.w;
  }
  s *= 0.08838834764831845f;  // 1/sqrt(128)
  alpha[t] = s;
  atomicMaxF(&amax[dst * HEADS + hd], s);
}

// ---------------- exp(alpha - max) and denominator ----------------
__global__ __launch_bounds__(256)
void edge_exp_k(const int* __restrict__ ei, float* __restrict__ alpha,
                const float* __restrict__ amax, float* __restrict__ den, int E)
{
  int t = blockIdx.x * 256 + threadIdx.x;
  if (t >= E * HEADS) return;
  int e = t >> 2, hd = t & 3;
  int dst = ei[E + e];
  float num = __expf(alpha[t] - amax[dst * HEADS + hd]);
  alpha[t] = num;
  atomicAdd(&den[dst * HEADS + hd], num);
}

// ---------------- weighted-value scatter (wave per (edge,head)) ----------------
__global__ __launch_bounds__(256)
void edge_agg_k(const int* __restrict__ ei, const float* __restrict__ alpha,
                const float* __restrict__ den, const float* __restrict__ v,
                float* __restrict__ agg, int E)
{
  int item = blockIdx.x * 8 + (threadIdx.x >> 5);
  if (item >= E * HEADS) return;
  int lane = threadIdx.x & 31;
  int e = item >> 2, hd = item & 3;
  int src = ei[e], dst = ei[E + e];
  float a = alpha[item] / den[dst * HEADS + hd];
  const float4 vv = *(const float4*)&v[(size_t)src * QKV + hd * DH + lane * 4];
  float* ap = &agg[(size_t)dst * QKV + hd * DH + lane * 4];
  atomicAdd(ap + 0, a * vv.x);
  atomicAdd(ap + 1, a * vv.y);
  atomicAdd(ap + 2, a * vv.z);
  atomicAdd(ap + 3, a * vv.w);
}

// ---------------- head-mean + beta gate + residual (wave per node) ----------------
__global__ __launch_bounds__(256)
void node_combine_k(const float* __restrict__ agg, const float* __restrict__ xr,
                    const float* __restrict__ wb, const float* __restrict__ g1p,
                    float* __restrict__ h, int n)
{
  int node = blockIdx.x * 8 + (threadIdx.x >> 5);
  if (node >= n) return;
  int lane = threadIdx.x & 31;
  int d = lane * 4;
  float4 o; o.x = 0.f; o.y = 0.f; o.z = 0.f; o.w = 0.f;
#pragma unroll
  for (int hd = 0; hd < HEADS; ++hd) {
    float4 tv = *(const float4*)&agg[(size_t)node * QKV + hd * DH + d];
    o.x += tv.x; o.y += tv.y; o.z += tv.z; o.w += tv.w;
  }
  o.x *= 0.25f; o.y *= 0.25f; o.z *= 0.25f; o.w *= 0.25f;
  float4 x  = *(const float4*)&xr[(size_t)node * H_DIM + d];
  float4 w0 = *(const float4*)&wb[d];
  float4 w1 = *(const float4*)&wb[H_DIM + d];
  float4 w2 = *(const float4*)&wb[2 * H_DIM + d];
  float part = o.x * w0.x + o.y * w0.y + o.z * w0.z + o.w * w0.w
             + x.x * w1.x + x.y * w1.y + x.z * w1.z + x.w * w1.w
             + (o.x - x.x) * w2.x + (o.y - x.y) * w2.y
             + (o.z - x.z) * w2.z + (o.w - x.w) * w2.w;
  part = warp_sum(part);
  float beta = 1.0f / (1.0f + __expf(-part));
  float g1 = g1p[0];
  float4 hv = *(float4*)&h[(size_t)node * H_DIM + d];
  hv.x += g1 * (beta * x.x + (1.0f - beta) * o.x);
  hv.y += g1 * (beta * x.y + (1.0f - beta) * o.y);
  hv.z += g1 * (beta * x.z + (1.0f - beta) * o.z);
  hv.w += g1 * (beta * x.w + (1.0f - beta) * o.w);
  *(float4*)&h[(size_t)node * H_DIM + d] = hv;
}

// ---------------- h += g2 * f ----------------
__global__ void resid_k(float* __restrict__ h, const float* __restrict__ f,
                        const float* __restrict__ g2p, int n) {
  int t = blockIdx.x * blockDim.x + threadIdx.x;
  if (t < n) h[t] += g2p[0] * f[t];
}

// ---------------- orchestration ----------------
extern "C" void kernel_launch(void* const* d_in, const int* in_sizes, int n_in,
                              void* d_out, int out_size, void* d_ws, size_t ws_size,
                              hipStream_t stream)
{
  const float* x      = (const float*)d_in[0];
  const int*   ei     = (const int*)  d_in[1];
  const float* Win    = (const float*)d_in[2];
  const float* b_in   = (const float*)d_in[3];
  const float* ln1_s  = (const float*)d_in[4];
  const float* ln1_b  = (const float*)d_in[5];
  const float* Wq     = (const float*)d_in[6];
  const float* bq     = (const float*)d_in[7];
  const float* Wk     = (const float*)d_in[8];
  const float* bk     = (const float*)d_in[9];
  const float* Wv     = (const float*)d_in[10];
  const float* bv     = (const float*)d_in[11];
  const float* Wskip  = (const float*)d_in[12];
  const float* bskip  = (const float*)d_in[13];
  const float* Wbeta  = (const float*)d_in[14];
  const float* ln2_s  = (const float*)d_in[15];
  const float* ln2_b  = (const float*)d_in[16];
  const float* W1     = (const float*)d_in[17];
  const float* b1     = (const float*)d_in[18];
  const float* W2     = (const float*)d_in[19];
  const float* b2     = (const float*)d_in[20];
  const float* g1     = (const float*)d_in[21];
  const float* g2     = (const float*)d_in[22];
  const float* lno_s  = (const float*)d_in[23];
  const float* lno_b  = (const float*)d_in[24];

  const int n = in_sizes[0] / F_DIM;
  const int E = in_sizes[1] / 2;

  // ---- workspace: f32 region then f16 region ----
  float* ws = (float*)d_ws;
  size_t o = 0;
  float* h     = ws + o; o += (size_t)n * H_DIM;
  float* xr    = ws + o; o += (size_t)n * H_DIM;
  float* q     = ws + o; o += (size_t)n * QKV;
  float* k     = ws + o; o += (size_t)n * QKV;
  float* v     = ws + o; o += (size_t)n * QKV;
  float* amax  = ws + o; o += (size_t)n * HEADS;
  float* den   = ws + o; o += (size_t)n * HEADS;   // den and agg contiguous
  float* agg   = ws + o; o += (size_t)n * QKV;
  float* alpha = ws + o; o += (size_t)E * HEADS;

  _Float16* hws = (_Float16*)(ws + o);
  size_t ho = 0;
  _Float16* x_h    = hws + ho; ho += (size_t)n * F_DIM;
  _Float16* hn_h   = hws + ho; ho += (size_t)n * H_DIM;
  _Float16* ffn1_h = hws + ho; ho += (size_t)n * QKV;
  _Float16* Win_h  = hws + ho; ho += F_DIM * H_DIM;
  _Float16* Wq_h   = hws + ho; ho += (size_t)LAYERS * H_DIM * QKV;
  _Float16* Wk_h   = hws + ho; ho += (size_t)LAYERS * H_DIM * QKV;
  _Float16* Wv_h   = hws + ho; ho += (size_t)LAYERS * H_DIM * QKV;
  _Float16* Wsk_h  = hws + ho; ho += (size_t)LAYERS * H_DIM * H_DIM;
  _Float16* W1_h   = hws + ho; ho += (size_t)LAYERS * H_DIM * 4 * H_DIM;
  _Float16* W2_h   = hws + ho; ho += (size_t)LAYERS * 4 * H_DIM * H_DIM;
  (void)ws_size; (void)n_in; (void)out_size;

  const dim3 blk(256);
  const int mtiles = (n + 127) / 128;
  const int node_blocks = (n + 7) / 8;
  const int eh = E * HEADS;
  const int eh_blocks = (eh + 255) / 256;
  const int eh_wave_blocks = (eh + 7) / 8;
  const float NEG_INF = -__builtin_huge_valf();

  // ---- one-time f16 conversions ----
  auto cvt = [&](const float* s, _Float16* d, size_t cnt) {
    cvt_h_k<<<(int)((cnt + 255) / 256), blk, 0, stream>>>(s, d, (int)cnt);
  };
  cvt(x, x_h, (size_t)n * F_DIM);
  cvt(Win, Win_h, F_DIM * H_DIM);
  cvt(Wq, Wq_h, (size_t)LAYERS * H_DIM * QKV);
  cvt(Wk, Wk_h, (size_t)LAYERS * H_DIM * QKV);
  cvt(Wv, Wv_h, (size_t)LAYERS * H_DIM * QKV);
  cvt(Wskip, Wsk_h, (size_t)LAYERS * H_DIM * H_DIM);
  cvt(W1, W1_h, (size_t)LAYERS * H_DIM * 4 * H_DIM);
  cvt(W2, W2_h, (size_t)LAYERS * 4 * H_DIM * H_DIM);

  // h = x @ Win + b_in
  gemm_f16_wmma<<<dim3(mtiles, H_DIM / 64), blk, 0, stream>>>(
      x_h, F_DIM, Win_h, H_DIM, b_in, h, nullptr, H_DIM, n, F_DIM, 0);

  for (int l = 0; l < LAYERS; ++l) {
    // hn = LN1(h)  (f16 out, feeds GEMMs)
    layernorm_h_k<<<node_blocks, blk, 0, stream>>>(h, ln1_s + l * H_DIM,
                                                   ln1_b + l * H_DIM, hn_h, n);
    // q/k/v = hn @ W + b
    gemm_f16_wmma<<<dim3(mtiles, QKV / 64), blk, 0, stream>>>(
        hn_h, H_DIM, Wq_h + (size_t)l * H_DIM * QKV, QKV, bq + (size_t)l * QKV,
        q, nullptr, QKV, n, H_DIM, 0);
    gemm_f16_wmma<<<dim3(mtiles, QKV / 64), blk, 0, stream>>>(
        hn_h, H_DIM, Wk_h + (size_t)l * H_DIM * QKV, QKV, bk + (size_t)l * QKV,
        k, nullptr, QKV, n, H_DIM, 0);
    gemm_f16_wmma<<<dim3(mtiles, QKV / 64), blk, 0, stream>>>(
        hn_h, H_DIM, Wv_h + (size_t)l * H_DIM * QKV, QKV, bv + (size_t)l * QKV,
        v, nullptr, QKV, n, H_DIM, 0);
    // xr = hn @ Wskip + bskip
    gemm_f16_wmma<<<dim3(mtiles, H_DIM / 64), blk, 0, stream>>>(
        hn_h, H_DIM, Wsk_h + (size_t)l * H_DIM * H_DIM, H_DIM,
        bskip + (size_t)l * H_DIM, xr, nullptr, H_DIM, n, H_DIM, 0);

    // init segment buffers
    fill_k<<<(n * HEADS + 255) / 256, blk, 0, stream>>>(amax, NEG_INF, n * HEADS);
    fill_k<<<(n * HEADS + n * QKV + 255) / 256, blk, 0, stream>>>(
        den, 0.0f, n * HEADS + n * QKV);

    // segment softmax + aggregate
    edge_logits_k<<<eh_blocks, blk, 0, stream>>>(ei, q, k, alpha, amax, E);
    edge_exp_k<<<eh_blocks, blk, 0, stream>>>(ei, alpha, amax, den, E);
    edge_agg_k<<<eh_wave_blocks, blk, 0, stream>>>(ei, alpha, den, v, agg, E);

    // beta gate + residual into h
    node_combine_k<<<node_blocks, blk, 0, stream>>>(
        agg, xr, Wbeta + (size_t)l * 3 * H_DIM, g1 + l, h, n);

    // FFN
    layernorm_h_k<<<node_blocks, blk, 0, stream>>>(h, ln2_s + l * H_DIM,
                                                   ln2_b + l * H_DIM, hn_h, n);
    gemm_f16_wmma<<<dim3(mtiles, (4 * H_DIM) / 64), blk, 0, stream>>>(
        hn_h, H_DIM, W1_h + (size_t)l * H_DIM * 4 * H_DIM, 4 * H_DIM,
        b1 + (size_t)l * 4 * H_DIM, nullptr, ffn1_h, 4 * H_DIM, n, H_DIM, 1);
    gemm_f16_wmma<<<dim3(mtiles, H_DIM / 64), blk, 0, stream>>>(
        ffn1_h, 4 * H_DIM, W2_h + (size_t)l * 4 * H_DIM * H_DIM, H_DIM,
        b2 + (size_t)l * H_DIM, xr, nullptr, H_DIM, n, 4 * H_DIM, 0);
    resid_k<<<(n * H_DIM + 255) / 256, blk, 0, stream>>>(h, xr, g2 + l,
                                                         n * H_DIM);
  }

  // final LN -> d_out
  layernorm_f_k<<<node_blocks, blk, 0, stream>>>(h, lno_s, lno_b, (float*)d_out, n);
}
